// CustomModel_18124761989675
// MI455X (gfx1250) — matrix operand order repaired and checked by reading, not compiled, not run on hardware
//
#include <hip/hip_runtime.h>
#include <hip/hip_bf16.h>

typedef _Float16 h16;
typedef __attribute__((ext_vector_type(16))) _Float16 v16h;
typedef __attribute__((ext_vector_type(8)))  float    v8f;

#define EPSV 1e-5f
#define CDIV(a,b) (((a)+(b)-1)/(b))

#if defined(__AMDGCN__)
#define USE_ASYNC_LDS 1
#else
#define USE_ASYNC_LDS 0
#endif

// ---------------------------------------------------------------- utilities
__global__ void k_f32_to_f16(const float* __restrict__ src, h16* __restrict__ dst, long n) {
  long i = (long)blockIdx.x * blockDim.x + threadIdx.x;
  if (i < n) dst[i] = (h16)src[i];
}

// mean over HW per (b, c). grid = (C, B), block = 256
__global__ void k_chan_mean(const h16* __restrict__ x, float* __restrict__ out, int hw) {
  __shared__ float red[256];
  long base = ((long)blockIdx.y * gridDim.x + blockIdx.x) * hw;
  float s = 0.f;
  for (int i = threadIdx.x; i < hw; i += 256) s += (float)x[base + i];
  red[threadIdx.x] = s; __syncthreads();
  for (int off = 128; off > 0; off >>= 1) {
    if (threadIdx.x < off) red[threadIdx.x] += red[threadIdx.x + off];
    __syncthreads();
  }
  if (threadIdx.x == 0) out[(long)blockIdx.y * gridDim.x + blockIdx.x] = red[0] / (float)hw;
}

// tiny attention MLP: pooled[B,cin] -> relu(fc1) -> fc2 -> softmax -> attn[B,8]
__global__ void k_attn_mlp(const float* __restrict__ pooled, const float* __restrict__ fc1,
                           const float* __restrict__ fc2, const float* __restrict__ fc2b,
                           float* __restrict__ attn, int B, int cin, int hid) {
  int b = blockIdx.x * blockDim.x + threadIdx.x;
  if (b >= B) return;
  float h[32];
  for (int j = 0; j < hid; ++j) {
    float s = 0.f;
    for (int c = 0; c < cin; ++c) s += fc1[j * cin + c] * pooled[b * cin + c];
    h[j] = s > 0.f ? s : 0.f;
  }
  float lg[8]; float mx = -1e30f;
  for (int k = 0; k < 8; ++k) {
    float s = fc2b[k];
    for (int j = 0; j < hid; ++j) s += fc2[k * hid + j] * h[j];
    lg[k] = s; mx = s > mx ? s : mx;
  }
  float den = 0.f;
  for (int k = 0; k < 8; ++k) { lg[k] = __expf(lg[k] - mx); den += lg[k]; }
  for (int k = 0; k < 8; ++k) attn[b * 8 + k] = lg[k] / den;
}

// per-sample aggregated dynamic weights: out[b,m,kpad] = sum_j attn[b,j]*dyw[j,m,kreal]
__global__ void k_agg_w(const float* __restrict__ dyw, const float* __restrict__ attn,
                        h16* __restrict__ out, int M, int kreal, int kpad) {
  long per = (long)M * kpad;
  long i = (long)blockIdx.x * blockDim.x + threadIdx.x;
  int b = blockIdx.y;
  if (i >= per) return;
  int m = (int)(i / kpad), k = (int)(i % kpad);
  float v = 0.f;
  if (k < kreal) {
    long wi = (long)m * kreal + k;
    long ws = (long)M * kreal;
    const float* a = attn + b * 8;
    #pragma unroll
    for (int j = 0; j < 8; ++j) v += a[j] * dyw[j * ws + wi];
  }
  out[(long)b * per + i] = (h16)v;
}

// -------------------------------------------- implicit-GEMM conv via WMMA
// Y[b] (M x N=OH*OW, f16) = Wm[b] (M x K, f16) * im2col(X[b]) (K x N)
// Block tile 64x64, K-stage 64 (2x32 unrolled). 256 threads = 8 wave32 waves;
// each wave owns a 16(M) x 32(N) sub-tile -> 4 v_wmma_f32_16x16x32_f16 per stage.
// A tile staged via global_load_async_to_lds_b128 (inline asm; ASYNCcnt-tracked).
// Requires: M % 64 == 0, K % 64 == 0 (zero-padded), 16B-aligned weight rows.
__launch_bounds__(256)
__global__ void k_conv_wmma(
    const h16* __restrict__ Wm, long wstride,
    const h16* __restrict__ X,  long xstride,
    h16* __restrict__ Y,        long ystride,
    int M, int K, int Kreal,
    int Cin, int H, int Wd,
    int KH, int KW, int stride, int pad,
    int OH, int OW,
    float* __restrict__ cSum, float* __restrict__ cSq, int doStats)
{
  __shared__ __align__(16) h16 As[64 * 64];    // [m][k]
  __shared__ __align__(16) h16 BsT[64 * 64];   // [n][k] (transposed: contiguous frag reads)
  __shared__ int sCi[64], sKh[64], sKw[64];
  __shared__ float sSum[64], sSq[64];

  const int tid  = threadIdx.x;
  const int wave = tid >> 5, lane = tid & 31;
  const int b = blockIdx.z;
  const int N = OH * OW;
  const int n_blk = blockIdx.x * 64;
  const int m_blk = blockIdx.y * 64;
  const int wm = (wave >> 1) * 16;   // wave M offset in tile
  const int wn = (wave & 1) * 32;    // wave N offset in tile
  const int khkw = KH * KW;

  if (doStats && tid < 64) { sSum[tid] = 0.f; sSq[tid] = 0.f; }

  const h16* Wp = Wm + (long)b * wstride;
  const h16* Xp = X  + (long)b * xstride;

  // per-thread im2col assignment: one n-row, 16 k's
  const int gn   = tid >> 2;             // 0..63 within tile
  const int kseg = (tid & 3) * 16;       // 0,16,32,48
  const int ngl  = n_blk + gn;
  int oyS = 0, oxS = 0; bool nok = ngl < N;
  if (nok) { int oy = ngl / OW, ox = ngl - oy * OW; oyS = oy * stride - pad; oxS = ox * stride - pad; }

  // A-tile copy assignment: one m-row, 16 contiguous k's (32 bytes)
  const int arow = tid >> 2;
  const int acol = (tid & 3) * 16;

  v8f c0 = {}; v8f c1 = {};

  for (int kb = 0; kb < K; kb += 64) {
    __syncthreads();   // previous stage fully consumed

    // ---- stage A tile via async DMA to LDS ----
    {
      const h16* src = Wp + (long)(m_blk + arow) * K + kb + acol;
      h16* dst = As + arow * 64 + acol;
#if USE_ASYNC_LDS
      // VDST = per-lane LDS byte offset (= low 32 bits of generic pointer),
      // VADDR = 64-bit global address; INST_OFFSET adds to both sides (ISA 15.18).
      unsigned ldsoff = (unsigned)(unsigned long long)(void*)dst;
      unsigned long long gaddr = (unsigned long long)(const void*)src;
      asm volatile("global_load_async_to_lds_b128 %0, %1, off"
                   :: "v"(ldsoff), "v"(gaddr) : "memory");
      asm volatile("global_load_async_to_lds_b128 %0, %1, off offset:16"
                   :: "v"(ldsoff), "v"(gaddr) : "memory");
#else
      #pragma unroll
      for (int j = 0; j < 16; ++j) dst[j] = src[j];
#endif
      if (kb + 64 < K) __builtin_prefetch(src + 64, 0, 3);
    }

    // ---- decode k -> (ci,kh,kw) once per stage ----
    if (tid < 64) {
      int kg = kb + tid;
      if (kg < Kreal) {
        int ci = kg / khkw; int r = kg - ci * khkw;
        sCi[tid] = ci; sKh[tid] = r / KW; sKw[tid] = r - (r / KW) * KW;
      } else {
        sCi[tid] = -1; sKh[tid] = 0; sKw[tid] = 0;
      }
    }
    __syncthreads();   // decode visible

    // ---- gather im2col B tile (transposed [n][k]) ----
    {
      h16* brow = BsT + gn * 64 + kseg;
      #pragma unroll
      for (int j = 0; j < 16; ++j) {
        int k = kseg + j;
        int ci = sCi[k];
        h16 v = (h16)0.f;
        if (nok && ci >= 0) {
          int iy = oyS + sKh[k];
          int ix = oxS + sKw[k];
          if (iy >= 0 && iy < H && ix >= 0 && ix < Wd)
            v = Xp[(long)ci * H * Wd + (long)iy * Wd + ix];
        }
        brow[j] = v;
      }
    }

#if USE_ASYNC_LDS
    asm volatile("s_wait_asynccnt 0" ::: "memory");  // own async A writes landed
#endif
    __syncthreads();   // all waves' A + B tiles visible

    // ---- 2 x (K=32) WMMA sub-steps ----
    #pragma unroll
    for (int ks = 0; ks < 2; ++ks) {
      const int kof = ks * 32;
      v16h a, b0v, b1v;
      {
        int mrow = wm + (lane & 15);
        int koff = (lane < 16) ? 0 : 16;       // A: lanes>=16 hold K 16..31 (contiguous)
        const h16* ar = As + mrow * 64 + kof;
        #pragma unroll
        for (int e = 0; e < 16; ++e) {
          int kk = koff + e;
          a[e] = ar[kk];
        }
        int ncol  = wn + (lane & 15);
        int kbase = (lane < 16) ? 0 : 16;      // B: N across lanes, K contiguous
        const h16* br0 = BsT + ncol * 64 + kof + kbase;
        const h16* br1 = BsT + (ncol + 16) * 64 + kof + kbase;
        #pragma unroll
        for (int e = 0; e < 16; ++e) b0v[e] = br0[e];
        #pragma unroll
        for (int e = 0; e < 16; ++e) b1v[e] = br1[e];
      }
      c0 = __builtin_amdgcn_wmma_f32_16x16x32_f16(false, a, false, b0v, (short)0, c0, false, false);
      c1 = __builtin_amdgcn_wmma_f32_16x16x32_f16(false, a, false, b1v, (short)0, c1, false, false);
    }
  }

  __syncthreads();
  // Epilogue: C/D layout -> VGPR r holds M = r + 8*(lane>=16), N = lane%16
  int mloc  = wm + (lane >= 16 ? 8 : 0);
  int nloc0 = n_blk + wn + (lane & 15);
  h16* Yp = Y + (long)b * ystride;
  #pragma unroll
  for (int r = 0; r < 8; ++r) {
    int m  = mloc + r;
    int gm = m_blk + m;
    float v0 = c0[r], v1 = c1[r];
    if (nloc0 < N) {
      Yp[(long)gm * N + nloc0] = (h16)v0;
      if (doStats) { atomicAdd(&sSum[m], v0); atomicAdd(&sSq[m], v0 * v0); }
    }
    if (nloc0 + 16 < N) {
      Yp[(long)gm * N + nloc0 + 16] = (h16)v1;
      if (doStats) { atomicAdd(&sSum[m], v1); atomicAdd(&sSq[m], v1 * v1); }
    }
  }
  if (doStats) {
    __syncthreads();
    if (tid < 64) {
      atomicAdd(&cSum[m_blk + tid], sSum[tid]);
      atomicAdd(&cSq[m_blk + tid],  sSq[tid]);
    }
  }
}

// ------------------------------------------------------ BN finalize / apply
__global__ void k_bn_finalize(const float* __restrict__ cSum, const float* __restrict__ cSq,
                              const float* __restrict__ g, const float* __restrict__ beta,
                              float* __restrict__ scale, float* __restrict__ shift,
                              int C, float invCount) {
  int c = blockIdx.x * blockDim.x + threadIdx.x;
  if (c >= C) return;
  float mean = cSum[c] * invCount;
  float var  = cSq[c] * invCount - mean * mean;
  float sc   = g[c] * rsqrtf(var + EPSV);
  scale[c] = sc;
  shift[c] = beta[c] - mean * sc;
}

// BN affine + 3x3 s2 p1 maxpool. grid = (ceil(OH*OW/256), C, B)
__global__ void k_bn_maxpool(const h16* __restrict__ x, h16* __restrict__ y,
                             const float* __restrict__ scale, const float* __restrict__ shift,
                             int C, int H, int W, int OH, int OW) {
  int n = blockIdx.x * 256 + threadIdx.x;
  if (n >= OH * OW) return;
  int c = blockIdx.y, b = blockIdx.z;
  int oy = n / OW, ox = n - oy * OW;
  const h16* xp = x + ((long)b * C + c) * H * W;
  float sc = scale[c], sh = shift[c];
  float m = -1e30f;
  for (int dy = 0; dy < 3; ++dy)
    for (int dx = 0; dx < 3; ++dx) {
      int iy = oy * 2 - 1 + dy, ix = ox * 2 - 1 + dx;
      if (iy >= 0 && iy < H && ix >= 0 && ix < W) {
        float v = (float)xp[iy * W + ix] * sc + sh;
        m = v > m ? v : m;
      }
    }
  y[(((long)b * C + c) * OH + oy) * OW + ox] = (h16)m;
}

// BN affine (+optional relu). grid = (ceil(HW/256), C, B)
__global__ void k_bn_apply(const h16* __restrict__ x, h16* __restrict__ y,
                           const float* __restrict__ scale, const float* __restrict__ shift,
                           int C, int HW, int relu) {
  int n = blockIdx.x * 256 + threadIdx.x;
  if (n >= HW) return;
  int c = blockIdx.y;
  long idx = ((long)blockIdx.z * C + c) * HW + n;
  float v = (float)x[idx] * scale[c] + shift[c];
  if (relu && v < 0.f) v = 0.f;
  y[idx] = (h16)v;
}

// relu(bn(l) + bn(r)). grid = (ceil(HW/256), C, B)
__global__ void k_res_join(const h16* __restrict__ l, const h16* __restrict__ r, h16* __restrict__ y,
                           const float* __restrict__ s1, const float* __restrict__ h1,
                           const float* __restrict__ s2, const float* __restrict__ h2,
                           int C, int HW) {
  int n = blockIdx.x * 256 + threadIdx.x;
  if (n >= HW) return;
  int c = blockIdx.y;
  long idx = ((long)blockIdx.z * C + c) * HW + n;
  float v = (float)l[idx] * s1[c] + h1[c] + (float)r[idx] * s2[c] + h2[c];
  y[idx] = (h16)(v > 0.f ? v : 0.f);
}

// feat = mean_hw(y) + bo ; out = feat @ fc_w^T + fc_b. grid = B, block = 256
__global__ void k_head(const h16* __restrict__ y, const float* __restrict__ bo,
                       const float* __restrict__ fcw, const float* __restrict__ fcb,
                       float* __restrict__ out) {
  __shared__ float feat[256];
  int b = blockIdx.x, t = threadIdx.x;
  const h16* yp = y + (long)b * 256 * 49;
  float s = 0.f;
  for (int i = 0; i < 49; ++i) s += (float)yp[t * 49 + i];
  feat[t] = s * (1.f / 49.f) + bo[t];
  __syncthreads();
  if (t < 50) {
    float o = fcb[t];
    for (int c = 0; c < 256; ++c) o += fcw[t * 256 + c] * feat[c];
    out[b * 50 + t] = o;
  }
}

// ---------------------------------------------------------------- launch
extern "C" void kernel_launch(void* const* d_in, const int* in_sizes, int n_in,
                              void* d_out, int out_size, void* d_ws, size_t ws_size,
                              hipStream_t stream) {
  (void)in_sizes; (void)n_in; (void)out_size; (void)ws_size;
  const float* imgs      = (const float*)d_in[0];
  const float* a1_fc1    = (const float*)d_in[1];
  const float* a1_fc2    = (const float*)d_in[2];
  const float* a1_fc2b   = (const float*)d_in[3];
  const float* dy1       = (const float*)d_in[4];
  const float* bn1_g     = (const float*)d_in[5];
  const float* bn1_b     = (const float*)d_in[6];
  const float* a2_fc1    = (const float*)d_in[7];
  const float* a2_fc2    = (const float*)d_in[8];
  const float* a2_fc2b   = (const float*)d_in[9];
  const float* dy2       = (const float*)d_in[10];
  const float* bn2_g     = (const float*)d_in[11];
  const float* bn2_b     = (const float*)d_in[12];
  const float* rw1       = (const float*)d_in[13];
  const float* rbn1_g    = (const float*)d_in[14];
  const float* rbn1_b    = (const float*)d_in[15];
  const float* rw2       = (const float*)d_in[16];
  const float* rbn2_g    = (const float*)d_in[17];
  const float* rbn2_b    = (const float*)d_in[18];
  const float* rws       = (const float*)d_in[19];
  const float* rbns_g    = (const float*)d_in[20];
  const float* rbns_b    = (const float*)d_in[21];
  // d_in[22]=wq, d_in[23]=wk unused: softmax over k sums to 1 -> out == v
  const float* wv        = (const float*)d_in[24];
  const float* wo        = (const float*)d_in[25];
  const float* bo        = (const float*)d_in[26];
  const float* fcw       = (const float*)d_in[27];
  const float* fcb       = (const float*)d_in[28];
  float* out = (float*)d_out;

  const int B = 64;
  char* ws = (char*)d_ws;
  size_t off = 0;
  auto alloc = [&](size_t bytes) -> void* {
    void* p = ws + off;
    off = (off + bytes + 255) & ~(size_t)255;
    return p;
  };

  h16* imgs_h = (h16*)alloc((size_t)B * 3 * 224 * 224 * 2);
  h16* x1     = (h16*)alloc((size_t)B * 64 * 112 * 112 * 2);
  h16* x1p    = (h16*)alloc((size_t)B * 64 * 56 * 56 * 2);
  h16* x2     = (h16*)alloc((size_t)B * 128 * 28 * 28 * 2);
  h16* x2p    = (h16*)alloc((size_t)B * 128 * 14 * 14 * 2);
  h16* w1agg  = (h16*)alloc((size_t)B * 64 * 192 * 2);
  h16* w2agg  = (h16*)alloc((size_t)B * 128 * 576 * 2);
  h16* rw1_h  = (h16*)alloc((size_t)256 * 1152 * 2);
  h16* rw2_h  = (h16*)alloc((size_t)256 * 2304 * 2);
  h16* rws_h  = (h16*)alloc((size_t)256 * 128 * 2);
  h16* wo_h   = (h16*)alloc((size_t)256 * 1024 * 2);
  h16* wv_h   = (h16*)alloc((size_t)1024 * 256 * 2);
  h16* Mmat   = (h16*)alloc((size_t)256 * 256 * 2);
  h16* l1     = (h16*)alloc((size_t)B * 256 * 49 * 2);
  h16* l1a    = (h16*)alloc((size_t)B * 256 * 49 * 2);
  h16* l2     = (h16*)alloc((size_t)B * 256 * 49 * 2);
  h16* rbuf   = (h16*)alloc((size_t)B * 256 * 49 * 2);
  h16* x3     = (h16*)alloc((size_t)B * 256 * 49 * 2);
  h16* ybuf   = (h16*)alloc((size_t)B * 256 * 49 * 2);
  float* pooled1 = (float*)alloc((size_t)B * 3 * 4);
  float* attn1   = (float*)alloc((size_t)B * 8 * 4);
  float* pooled2 = (float*)alloc((size_t)B * 64 * 4);
  float* attn2   = (float*)alloc((size_t)B * 8 * 4);
  float* stats   = (float*)alloc((size_t)5 * 512 * 4);  // per stage: sum[256], sq[256]
  float* scales  = (float*)alloc((size_t)5 * 512 * 4);  // per stage: scale[256], shift[256]

  float* sum1 = stats + 0 * 512; float* sq1 = sum1 + 256;
  float* sum2 = stats + 1 * 512; float* sq2 = sum2 + 256;
  float* sum3 = stats + 2 * 512; float* sq3 = sum3 + 256;
  float* sum4 = stats + 3 * 512; float* sq4 = sum4 + 256;
  float* sum5 = stats + 4 * 512; float* sq5 = sum5 + 256;
  float* sc1 = scales + 0 * 512; float* sh1 = sc1 + 256;
  float* sc2 = scales + 1 * 512; float* sh2 = sc2 + 256;
  float* sc3 = scales + 2 * 512; float* sh3 = sc3 + 256;
  float* sc4 = scales + 3 * 512; float* sh4 = sc4 + 256;
  float* sc5 = scales + 4 * 512; float* sh5 = sc5 + 256;

  (void)hipMemsetAsync(stats, 0, 5 * 512 * sizeof(float), stream);

  // f32 -> f16 conversions (inputs & static weights)
  { long n = (long)B * 3 * 224 * 224; k_f32_to_f16<<<CDIV(n, 256), 256, 0, stream>>>(imgs, imgs_h, n); }
  { long n = 256 * 1152; k_f32_to_f16<<<CDIV(n, 256), 256, 0, stream>>>(rw1, rw1_h, n); }
  { long n = 256 * 2304; k_f32_to_f16<<<CDIV(n, 256), 256, 0, stream>>>(rw2, rw2_h, n); }
  { long n = 256 * 128;  k_f32_to_f16<<<CDIV(n, 256), 256, 0, stream>>>(rws, rws_h, n); }
  { long n = 256 * 1024; k_f32_to_f16<<<CDIV(n, 256), 256, 0, stream>>>(wo, wo_h, n); }
  { long n = 1024 * 256; k_f32_to_f16<<<CDIV(n, 256), 256, 0, stream>>>(wv, wv_h, n); }

  // ---- stage 1: dynamic conv (3 -> 64, 7x7 s2 p3), K padded 147 -> 192 ----
  k_chan_mean<<<dim3(3, B), 256, 0, stream>>>(imgs_h, pooled1, 224 * 224);
  k_attn_mlp<<<1, 64, 0, stream>>>(pooled1, a1_fc1, a1_fc2, a1_fc2b, attn1, B, 3, 8);
  k_agg_w<<<dim3(CDIV(64 * 192, 256), B), 256, 0, stream>>>(dy1, attn1, w1agg, 64, 147, 192);
  k_conv_wmma<<<dim3(CDIV(112 * 112, 64), 1, B), 256, 0, stream>>>(
      w1agg, 64L * 192, imgs_h, 3L * 224 * 224, x1, 64L * 112 * 112,
      64, 192, 147, 3, 224, 224, 7, 7, 2, 3, 112, 112, sum1, sq1, 1);
  k_bn_finalize<<<1, 64, 0, stream>>>(sum1, sq1, bn1_g, bn1_b, sc1, sh1, 64, 1.f / (64.f * 112 * 112));
  k_bn_maxpool<<<dim3(CDIV(56 * 56, 256), 64, B), 256, 0, stream>>>(x1, x1p, sc1, sh1, 64, 112, 112, 56, 56);

  // ---- stage 2: dynamic conv (64 -> 128, 3x3 s2 p1), K = 576 ----
  k_chan_mean<<<dim3(64, B), 256, 0, stream>>>(x1p, pooled2, 56 * 56);
  k_attn_mlp<<<1, 64, 0, stream>>>(pooled2, a2_fc1, a2_fc2, a2_fc2b, attn2, B, 64, 17);
  k_agg_w<<<dim3(CDIV(128 * 576, 256), B), 256, 0, stream>>>(dy2, attn2, w2agg, 128, 576, 576);
  k_conv_wmma<<<dim3(CDIV(28 * 28, 64), 2, B), 256, 0, stream>>>(
      w2agg, 128L * 576, x1p, 64L * 56 * 56, x2, 128L * 28 * 28,
      128, 576, 576, 64, 56, 56, 3, 3, 2, 1, 28, 28, sum2, sq2, 1);
  k_bn_finalize<<<1, 128, 0, stream>>>(sum2, sq2, bn2_g, bn2_b, sc2, sh2, 128, 1.f / (64.f * 28 * 28));
  k_bn_maxpool<<<dim3(1, 128, B), 256, 0, stream>>>(x2, x2p, sc2, sh2, 128, 28, 28, 14, 14);

  // ---- residual block ----
  k_conv_wmma<<<dim3(1, 4, B), 256, 0, stream>>>(
      rw1_h, 0, x2p, 128L * 196, l1, 256L * 49,
      256, 1152, 1152, 128, 14, 14, 3, 3, 2, 1, 7, 7, sum3, sq3, 1);
  k_bn_finalize<<<1, 256, 0, stream>>>(sum3, sq3, rbn1_g, rbn1_b, sc3, sh3, 256, 1.f / (64.f * 49));
  k_bn_apply<<<dim3(1, 256, B), 256, 0, stream>>>(l1, l1a, sc3, sh3, 256, 49, 1);
  k_conv_wmma<<<dim3(1, 4, B), 256, 0, stream>>>(
      rw2_h, 0, l1a, 256L * 49, l2, 256L * 49,
      256, 2304, 2304, 256, 7, 7, 3, 3, 1, 1, 7, 7, sum4, sq4, 1);
  k_conv_wmma<<<dim3(1, 4, B), 256, 0, stream>>>(
      rws_h, 0, x2p, 128L * 196, rbuf, 256L * 49,
      256, 128, 128, 128, 14, 14, 1, 1, 2, 0, 7, 7, sum5, sq5, 1);
  k_bn_finalize<<<1, 256, 0, stream>>>(sum4, sq4, rbn2_g, rbn2_b, sc4, sh4, 256, 1.f / (64.f * 49));
  k_bn_finalize<<<1, 256, 0, stream>>>(sum5, sq5, rbns_g, rbns_b, sc5, sh5, 256, 1.f / (64.f * 49));
  k_res_join<<<dim3(1, 256, B), 256, 0, stream>>>(l2, rbuf, x3, sc4, sh4, sc5, sh5, 256, 49);

  // ---- self-attention (degenerate: out == v) ----
  // Mmat = wo @ wv : treat wv [1024,256] as 1024-channel 16x16 "image", 1x1 conv by wo.
  k_conv_wmma<<<dim3(4, 4, 1), 256, 0, stream>>>(
      wo_h, 0, wv_h, 0, Mmat, 0,
      256, 1024, 1024, 1024, 16, 16, 1, 1, 1, 0, 16, 16, nullptr, nullptr, 0);
  // y = Mmat @ x3 per pixel (1x1 conv, 256 -> 256)
  k_conv_wmma<<<dim3(1, 4, B), 256, 0, stream>>>(
      Mmat, 0, x3, 256L * 49, ybuf, 256L * 49,
      256, 256, 256, 256, 7, 7, 1, 1, 1, 0, 7, 7, nullptr, nullptr, 0);

  // ---- head: global mean + bias + fc ----
  k_head<<<B, 256, 0, stream>>>(ybuf, bo, fcw, fcb, out);
}
